// GCNScoreMatchingDenoisingModel_13116830122442
// MI455X (gfx1250) — compile-verified, matrix operand order checked by reading, not run on hardware
//
#include <hip/hip_runtime.h>

typedef __attribute__((ext_vector_type(16))) _Float16 v16h;
typedef __attribute__((ext_vector_type(8)))  _Float16 v8h;
typedef __attribute__((ext_vector_type(8)))  float    v8f;

static __device__ __forceinline__ int gtid() {
  return blockIdx.x * blockDim.x + threadIdx.x;
}

// deg starts at 1.0 (self-loop contribution to in-degree)
__global__ void k_init_deg(float* __restrict__ deg, int N) {
  int i = gtid();
  if (i < N) deg[i] = 1.0f;
}

// in-degree over edge destinations
__global__ void k_deg_edges(const int* __restrict__ dst, float* __restrict__ deg, int E) {
  int e = gtid();
  if (e < E) atomicAdd(&deg[dst[e]], 1.0f);
}

// dis = rsqrt(deg) in place; also write self-loop term of layer-1 aggregation:
// aggx[i] = x[i] * dis[i]^2   (fully initializes aggx, no zeroing needed)
__global__ void k_rsqrt_self(const float* __restrict__ x, float* __restrict__ dis,
                             float* __restrict__ aggx, int N) {
  int i = gtid();
  if (i >= N) return;
  float r = rsqrtf(dis[i]);       // deg >= 1 always (self loop)
  dis[i] = r;
  float nn = r * r;
  float4 xv = reinterpret_cast<const float4*>(x)[i];
  float4 o = make_float4(xv.x * nn, xv.y * nn, xv.z * nn, xv.w * nn);
  reinterpret_cast<float4*>(aggx)[i] = o;
}

// Layer-1 aggregation on raw 4-wide features (A(XW) == (AX)W):
// aggx[dst] += x[src] * norm  -> only 4 floats/edge instead of 64
__global__ void k_agg1(const int* __restrict__ src, const int* __restrict__ dst,
                       const float* __restrict__ x, const float* __restrict__ dis,
                       float* __restrict__ aggx, int E) {
  int e = gtid();
  if (e >= E) return;
  int s = src[e], d = dst[e];
  float n = dis[s] * dis[d];
  float4 xv = reinterpret_cast<const float4*>(x)[s];
  float* ap = aggx + (size_t)d * 4;
  atomicAdd(ap + 0, xv.x * n);
  atomicAdd(ap + 1, xv.y * n);
  atomicAdd(ap + 2, xv.z * n);
  atomicAdd(ap + 3, xv.w * n);
}

// h1 = relu(aggx @ W1 + b1)  (K=4, scalar FMAs); also init layer-2 accumulator
// with the self-loop term: r2[i] = h1[i] * dis[i]^2
__global__ void k_h1(const float* __restrict__ aggx, const float* __restrict__ W1,
                     const float* __restrict__ b1, const float* __restrict__ dis,
                     float* __restrict__ h1, float* __restrict__ r2, int N) {
  int i = gtid();
  if (i >= N * 64) return;
  int n = i >> 6, f = i & 63;
  float4 av = reinterpret_cast<const float4*>(aggx)[n];
  float acc = b1[f];
  acc += av.x * W1[f];
  acc += av.y * W1[64 + f];
  acc += av.z * W1[128 + f];
  acc += av.w * W1[192 + f];
  float h = fmaxf(acc, 0.0f);
  h1[i] = h;
  float r = dis[n];
  r2[i] = h * r * r;
}

// Layer-2 aggregation: 16 threads per edge, each handles 4 features (float4).
// r2[dst] += h1[src] * norm. Features are L2-resident; indices stream from HBM.
__global__ void k_agg2(const int* __restrict__ src, const int* __restrict__ dst,
                       const float* __restrict__ h1, const float* __restrict__ dis,
                       float* __restrict__ r2, int E) {
  long long t = (long long)blockIdx.x * blockDim.x + threadIdx.x;
  int e  = (int)(t >> 4);
  int f4 = (int)(t & 15);
  if (e >= E) return;
  if (f4 == 0) {
    __builtin_prefetch(src + e + 8192, 0, 1);  // global_prefetch_b8 on index stream
    __builtin_prefetch(dst + e + 8192, 0, 1);
  }
  int s = src[e], d = dst[e];
  float n = dis[s] * dis[d];
  float4 v = reinterpret_cast<const float4*>(h1 + (size_t)s * 64)[f4];
  float* rp = r2 + (size_t)d * 64 + f4 * 4;
  atomicAdd(rp + 0, v.x * n);
  atomicAdd(rp + 1, v.y * n);
  atomicAdd(rp + 2, v.z * n);
  atomicAdd(rp + 3, v.w * n);
}

// Fused: h2 = relu(r2 @ W2 + b2); out = h2 @ Wf + bf, written directly.
// Block = 128 threads = 4 waves; each wave computes a 16-row x 64-col tile via
// 8x V_WMMA_F32_16X16X32_F16, then reduces its in-register h2 tile against
// Wf[64,2] with 16-lane shfl_xor reductions — h2 never touches memory.
// W2 is held TRANSPOSED in LDS as f16 so each B fragment is two ds_load_b128.
// r2 is padded to a multiple of 64 rows -> no clamp, EXEC all-1s throughout.
__global__ void __launch_bounds__(128)
k_gemm2_out(const float* __restrict__ r2, const float* __restrict__ W2,
            const float* __restrict__ b2, const float* __restrict__ Wf,
            const float* __restrict__ bf, float* __restrict__ out, int N) {
  __shared__ _Float16 w2t[64 * 64];           // w2t[col*64 + k] = W2[k][col]
  for (int idx = threadIdx.x; idx < 64 * 64; idx += 128) {
    int col = idx >> 6, k = idx & 63;
    w2t[idx] = (_Float16)W2[k * 64 + col];
  }
  __syncthreads();

  int wave  = threadIdx.x >> 5;
  int lane  = threadIdx.x & 31;
  int rbase = blockIdx.x * 64 + wave * 16;
  int lrow  = lane & 15;
  int hi    = lane >> 4;      // 0: lanes 0-15, 1: lanes 16-31
  int kb    = hi * 8;

  const float* ar = r2 + (size_t)(rbase + lrow) * 64;   // padded: always in-bounds
  const _Float16* wt = w2t + (size_t)lrow * 64;         // per-lane col base (ct adds 16*64)

  v8f zero = {0, 0, 0, 0, 0, 0, 0, 0};
  v8f acc[4];
  acc[0] = zero; acc[1] = zero; acc[2] = zero; acc[3] = zero;

#pragma unroll
  for (int kc = 0; kc < 64; kc += 32) {
    // A fragment: lanes 0-15 hold K {kc+0..7, kc+16..23}; lanes 16-31 shift by 8
    v16h a;
#pragma unroll
    for (int j = 0; j < 8; ++j) {
      a[j]     = (_Float16)ar[kc + kb + j];
      a[8 + j] = (_Float16)ar[kc + kb + 16 + j];
    }
#pragma unroll
    for (int ct = 0; ct < 4; ++ct) {
      const _Float16* wc = wt + (size_t)ct * 16 * 64;   // col = ct*16 + lrow
      v8h blo = *reinterpret_cast<const v8h*>(wc + kc + kb);        // ds_load_b128
      v8h bhi = *reinterpret_cast<const v8h*>(wc + kc + kb + 16);   // ds_load_b128
      v16h b = __builtin_shufflevector(blo, bhi, 0, 1, 2, 3, 4, 5, 6, 7,
                                       8, 9, 10, 11, 12, 13, 14, 15);
      acc[ct] = __builtin_amdgcn_wmma_f32_16x16x32_f16(
          false, a, false, b, (short)0, acc[ct], false, false);
    }
  }

  // Per-lane epilogue constants (col = ct*16 + lrow)
  float bias[4], wf0[4], wf1[4];
#pragma unroll
  for (int ct = 0; ct < 4; ++ct) {
    int col = ct * 16 + lrow;
    bias[ct] = b2[col];
    wf0[ct]  = Wf[col * 2 + 0];
    wf1[ct]  = Wf[col * 2 + 1];
  }
  float bf0 = bf[0], bf1 = bf[1];

  // C/D layout: VGPR v, lanes 0-15 -> M=v, lanes 16-31 -> M=v+8; N = lane%16.
  // For each of the wave's 16 rows, h2 row lives across 16 lanes x 4 regs:
  // apply bias+relu, project onto Wf, reduce across the 16-lane half, store.
#pragma unroll
  for (int v = 0; v < 8; ++v) {
    float p0 = 0.0f, p1 = 0.0f;
#pragma unroll
    for (int ct = 0; ct < 4; ++ct) {
      float h = fmaxf(acc[ct][v] + bias[ct], 0.0f);
      p0 += h * wf0[ct];
      p1 += h * wf1[ct];
    }
#pragma unroll
    for (int off = 8; off >= 1; off >>= 1) {
      p0 += __shfl_xor(p0, off, 16);
      p1 += __shfl_xor(p1, off, 16);
    }
    int row = rbase + v + hi * 8;
    if (lrow == 0 && row < N) {
      out[row * 2 + 0] = p0 + bf0;
      out[row * 2 + 1] = p1 + bf1;
    }
  }
}

extern "C" void kernel_launch(void* const* d_in, const int* in_sizes, int n_in,
                              void* d_out, int out_size, void* d_ws, size_t ws_size,
                              hipStream_t stream) {
  const float* x  = (const float*)d_in[0];
  const int*   ei = (const int*)d_in[1];   // edge_index, [2, E] flat
  const float* W1 = (const float*)d_in[2];
  const float* b1 = (const float*)d_in[3];
  const float* W2 = (const float*)d_in[4];
  const float* b2 = (const float*)d_in[5];
  const float* Wf = (const float*)d_in[6];
  const float* bf = (const float*)d_in[7];
  float* out = (float*)d_out;

  int N = in_sizes[0] / 4;
  int E = in_sizes[1] / 2;
  int Npad = ((N + 63) / 64) * 64;          // r2 padded so WMMA tiles need no clamp
  const int* src = ei;
  const int* dst = ei + E;

  // workspace layout (floats): dis[N] | aggx[4N] | h1[64N] | r2[64*Npad]
  float* ws   = (float*)d_ws;
  float* dis  = ws;
  float* aggx = dis + N;
  float* h1   = aggx + (size_t)4 * N;
  float* r2   = h1 + (size_t)64 * N;

  const int TB = 256;
  k_init_deg<<<(N + TB - 1) / TB, TB, 0, stream>>>(dis, N);
  k_deg_edges<<<(E + TB - 1) / TB, TB, 0, stream>>>(dst, dis, E);
  k_rsqrt_self<<<(N + TB - 1) / TB, TB, 0, stream>>>(x, dis, aggx, N);
  k_agg1<<<(E + TB - 1) / TB, TB, 0, stream>>>(src, dst, x, dis, aggx, E);
  long long t1 = (long long)N * 64;
  k_h1<<<(unsigned)((t1 + TB - 1) / TB), TB, 0, stream>>>(aggx, W1, b1, dis, h1, r2, N);
  long long t2 = (long long)E * 16;
  k_agg2<<<(unsigned)((t2 + TB - 1) / TB), TB, 0, stream>>>(src, dst, h1, dis, r2, E);
  k_gemm2_out<<<Npad / 64, 128, 0, stream>>>(r2, W2, b2, Wf, bf, out, N);
}
